// LigerFusedLinearKTOLoss_80401787781833
// MI455X (gfx1250) — matrix operand order repaired and checked by reading, not compiled, not run on hardware
//
#include <hip/hip_runtime.h>
#include <hip/hip_bf16.h>
#include <cstddef>

// Problem constants (from reference)
#define BB   4
#define TT   512
#define HH   4096
#define VV   32000
#define BT   (BB * TT)      // 2048 rows
#define VT   (VV / 128)     // 250 V-tiles of 128 columns
#define MT2  (BT / 32)      // 64 M-tiles of 32 rows
#define IGNORE_INDEX (-100)
#define BETA 0.1f

typedef __attribute__((ext_vector_type(16))) __bf16 v16bf;
typedef __attribute__((ext_vector_type(8)))  float  v8f;

__device__ __forceinline__ __bf16 f2bf(float f) { return (__bf16)f; }

__device__ __forceinline__ unsigned pack2bf(float a, float b) {
    unsigned short lo = __builtin_bit_cast(unsigned short, f2bf(a));
    unsigned short hi = __builtin_bit_cast(unsigned short, f2bf(b));
    return (unsigned)lo | ((unsigned)hi << 16);
}

union AFrag {
    uint4 u[2];
    v16bf v;
};

__device__ __forceinline__ v16bf cvt16(const float4 q0, const float4 q1,
                                       const float4 q2, const float4 q3) {
    v16bf r;
    r[0]  = f2bf(q0.x); r[1]  = f2bf(q0.y); r[2]  = f2bf(q0.z); r[3]  = f2bf(q0.w);
    r[4]  = f2bf(q1.x); r[5]  = f2bf(q1.y); r[6]  = f2bf(q1.z); r[7]  = f2bf(q1.w);
    r[8]  = f2bf(q2.x); r[9]  = f2bf(q2.y); r[10] = f2bf(q2.z); r[11] = f2bf(q2.w);
    r[12] = f2bf(q3.x); r[13] = f2bf(q3.y); r[14] = f2bf(q3.z); r[15] = f2bf(q3.w);
    return r;
}

// -------------------------------------------------------------------------
// Kernel 1: fused GEMM (bf16 WMMA) + bias + tile-local logsumexp partials
// grid = (MT2, VT, 2 models), block = 256 threads (8 waves).
// Block tile: 32 rows x 128 vocab cols. Each wave owns a 16-col strip and
// computes two 16x16 WMMA tiles (M sub-tiles) sharing each B fragment.
// K loop steps by 64 (two K-slabs per LDS stage), LDS A is double-buffered
// so there is ONE __syncthreads per iteration.
// -------------------------------------------------------------------------
__global__ void __launch_bounds__(256)
kto_gemm_lse(const float* __restrict__ x0,  const float* __restrict__ w0,
             const float* __restrict__ b0,
             const float* __restrict__ x1,  const float* __restrict__ w1,
             const float* __restrict__ b1,
             const int*   __restrict__ target,
             float* __restrict__ ws_pmax,   // [2][BT][VT]
             float* __restrict__ ws_psum,   // [2][BT][VT]
             float* __restrict__ ws_tgt)    // [2][BT]
{
    __shared__ unsigned lds_a[2][32][32];  // ping-pong: 32 rows x 64 K as bf16 pairs
    __shared__ float    ltile[32][128];    // block logits tile
    __shared__ float    red[32][8];
    __shared__ float    rowmax_s[32];

    const int model = blockIdx.z;
    const float* __restrict__ x    = model ? x1 : x0;
    const float* __restrict__ w    = model ? w1 : w0;
    const float* __restrict__ bias = model ? b1 : b0;

    const int tid  = threadIdx.x;
    const int wave = tid >> 5;
    const int lane = tid & 31;

    const int row0 = blockIdx.x * 32;          // M-tile base (fastest grid dim -> W L2 reuse)
    const int col0 = blockIdx.y * 128;         // V-tile base

    const int nlo   = lane & 15;               // output column within wave strip
    const int khalf = (lane >> 4);             // 0/1: which K-half this lane owns (A and B)
    const int colw  = col0 + wave * 16 + nlo;  // vocab column owned by this lane
    const float* __restrict__ wrow = w + (size_t)colw * HH + (khalf << 4);

    v8f c[2] = {{}, {}};

    // --- staging: 32 rows x 64 K = 2048 floats, 8 consecutive floats/thread ---
    const int sr  = tid >> 3;                  // row 0..31
    const int skk = (tid & 7) * 8;             // K offset 0..56
    const float* __restrict__ xrow = x + (size_t)(row0 + sr) * HH + skk;
    const int spair = skk >> 1;                // uint pair index (multiple of 4)

    // --- A-fragment pair offsets (contiguous K order, same as B) ---
    // lane<16: K {0..15}; lane>=16: K {16..31} (per 32-K slab)
    const int arow = lane & 15;
    const int apair = khalf * 8;

    // prologue: stage slab 0 into buffer 0
    {
        const float2 xa = *(const float2*)(xrow + 0);
        const float2 xb = *(const float2*)(xrow + 2);
        const float2 xc = *(const float2*)(xrow + 4);
        const float2 xd = *(const float2*)(xrow + 6);
        uint4 st;
        st.x = pack2bf(xa.x, xa.y);
        st.y = pack2bf(xb.x, xb.y);
        st.z = pack2bf(xc.x, xc.y);
        st.w = pack2bf(xd.x, xd.y);
        *(uint4*)&lds_a[0][sr][spair] = st;
    }
    __syncthreads();

    const int NITER = HH / 64;   // 64 iterations
    for (int it = 0; it < NITER; ++it) {
        const int k0  = it * 64;
        const int buf = it & 1;

        // --- stage NEXT slab into the other buffer (overlaps with compute) ---
        if (it + 1 < NITER) {
            const float* __restrict__ xs = xrow + k0 + 64;
            const float2 xa = *(const float2*)(xs + 0);
            const float2 xb = *(const float2*)(xs + 2);
            const float2 xc = *(const float2*)(xs + 4);
            const float2 xd = *(const float2*)(xs + 6);
            uint4 st;
            st.x = pack2bf(xa.x, xa.y);
            st.y = pack2bf(xb.x, xb.y);
            st.z = pack2bf(xc.x, xc.y);
            st.w = pack2bf(xd.x, xd.y);
            *(uint4*)&lds_a[buf ^ 1][sr][spair] = st;
        }

        if (k0 + 192 < HH) __builtin_prefetch(wrow + k0 + 192, 0, 3);

        // --- two K-slabs of 32; B fragment shared by both M sub-tiles ---
        #pragma unroll
        for (int ks = 0; ks < 2; ++ks) {
            // B fragment: 16 contiguous f32 of one W row -> bf16
            const float4* __restrict__ wp = (const float4*)(wrow + k0 + ks * 32);
            const v16bf bfv = cvt16(wp[0], wp[1], wp[2], wp[3]);

            #pragma unroll
            for (int ms = 0; ms < 2; ++ms) {
                AFrag af;
                const int pr = ks * 16 + apair;
                af.u[0] = *(const uint4*)&lds_a[buf][ms * 16 + arow][pr];
                af.u[1] = *(const uint4*)&lds_a[buf][ms * 16 + arow][pr + 4];
                c[ms] = __builtin_amdgcn_wmma_f32_16x16x32_bf16(
                            false, af.v, false, bfv,
                            (short)0, c[ms], false, false);
            }
        }

        __syncthreads();   // staged buf^1 visible; reads of buf done before overwrite
    }

    // --- epilogue: bias add, spill 32x128 tile to LDS ---
    {
        const float bv = bias[colw];
        const int mbase = khalf * 8;   // C/D layout: VGPR j -> M = j (+8 for lanes 16..31)
        #pragma unroll
        for (int ms = 0; ms < 2; ++ms)
            #pragma unroll
            for (int j = 0; j < 8; ++j)
                ltile[ms * 16 + mbase + j][wave * 16 + nlo] = c[ms][j] + bv;
    }
    __syncthreads();

    // --- per-row max over 128 columns (8 threads per row, 16 cols each) ---
    const int rrow = tid >> 3;
    const int rc   = tid & 7;
    {
        float m = -__builtin_inff();
        #pragma unroll
        for (int i = 0; i < 16; ++i) m = fmaxf(m, ltile[rrow][rc + 8 * i]);
        red[rrow][rc] = m;
    }
    __syncthreads();
    if (rc == 0) {
        float mm = red[rrow][0];
        #pragma unroll
        for (int i = 1; i < 8; ++i) mm = fmaxf(mm, red[rrow][i]);
        rowmax_s[rrow] = mm;
    }
    __syncthreads();

    // --- per-row sum(exp(x - max)) ---
    {
        const float rm = rowmax_s[rrow];
        float s = 0.f;
        #pragma unroll
        for (int i = 0; i < 16; ++i) s += __expf(ltile[rrow][rc + 8 * i] - rm);
        red[rrow][rc] = s;
    }
    __syncthreads();
    if (rc == 0) {
        float ss = 0.f;
        #pragma unroll
        for (int i = 0; i < 8; ++i) ss += red[rrow][i];
        const size_t idx = ((size_t)model * BT + row0 + rrow) * VT + blockIdx.y;
        ws_pmax[idx] = rowmax_s[rrow];
        ws_psum[idx] = ss;
    }

    // --- target logit gather (exactly one V-tile block owns each target) ---
    if (tid < 32) {
        const int r  = row0 + tid;
        int tg = target[r];
        if (tg < 0) tg = 0;                    // clip, as in reference
        if (tg >= col0 && tg < col0 + 128)
            ws_tgt[(size_t)model * BT + r] = ltile[tid][tg - col0];
    }
}

// -------------------------------------------------------------------------
// Kernel 2: combine streaming-softmax partials -> per-token log p(target)
// -------------------------------------------------------------------------
__global__ void __launch_bounds__(256)
kto_combine(const float* __restrict__ ws_pmax,
            const float* __restrict__ ws_psum,
            const float* __restrict__ ws_tgt,
            float* __restrict__ ws_logp)      // [2][BT]
{
    const int gid = blockIdx.x * blockDim.x + threadIdx.x;
    if (gid >= 2 * BT) return;
    const int model = gid >> 11;          // BT == 2048 == 1<<11
    const int r     = gid & (BT - 1);

    const float* pm = ws_pmax + ((size_t)model * BT + r) * VT;
    const float* ps = ws_psum + ((size_t)model * BT + r) * VT;

    float g = -__builtin_inff();
    for (int i = 0; i < VT; ++i) g = fmaxf(g, pm[i]);
    float s = 0.f;
    for (int i = 0; i < VT; ++i) s += ps[i] * __expf(pm[i] - g);
    const float lse = g + __logf(s);
    ws_logp[(size_t)model * BT + r] = ws_tgt[(size_t)model * BT + r] - lse;
}

// -------------------------------------------------------------------------
// Kernel 3: masked per-batch averages + KTO loss
// -------------------------------------------------------------------------
__global__ void __launch_bounds__(64)
kto_finalize(const float* __restrict__ ws_logp,
             const int*   __restrict__ target,
             const int*   __restrict__ pref,
             const float* __restrict__ kl,
             float* __restrict__ out)
{
    __shared__ float ssum[2][BB];
    __shared__ float scnt[BB];
    const int t = threadIdx.x;
    if (t < 2 * BB) {
        const int model = t >> 2, b = t & 3;
        float s = 0.f, c = 0.f;
        for (int i = 0; i < TT; ++i) {
            const int r = b * TT + i;
            if (target[r] != IGNORE_INDEX) {
                s += ws_logp[(size_t)model * BT + r];
                c += 1.f;
            }
        }
        ssum[model][b] = s;
        if (model == 0) scnt[b] = c;
    }
    __syncthreads();
    if (t == 0) {
        const float kl0 = kl[0];
        float acc = 0.f;
        for (int b = 0; b < BB; ++b) {
            const float d  = fmaxf(scnt[b], 1.f);
            const float lr = ssum[0][b] / d - ssum[1][b] / d;
            const float mult = pref[b] ? 1.f : -1.f;
            const float z = BETA * (lr - kl0) * mult;
            acc += 1.f - 1.f / (1.f + __expf(-z));
        }
        out[0] = acc / (float)BB;
    }
}

// -------------------------------------------------------------------------
extern "C" void kernel_launch(void* const* d_in, const int* in_sizes, int n_in,
                              void* d_out, int out_size, void* d_ws, size_t ws_size,
                              hipStream_t stream) {
    const float* x  = (const float*)d_in[0];   // _input [B,T,H]
    const float* w  = (const float*)d_in[1];   // weight [V,H]
    const float* bs = (const float*)d_in[2];   // bias [V]
    const int*   tg = (const int*)d_in[3];     // target [B,T]
    const int*   pl = (const int*)d_in[4];     // preference_labels [B]
    const float* rx = (const float*)d_in[5];   // ref_input
    const float* rw = (const float*)d_in[6];   // ref_weight
    const float* rb = (const float*)d_in[7];   // ref_bias
    const float* kl = (const float*)d_in[8];   // kl [1]

    // workspace layout (floats): pmax | psum | tgt | logp  (~8.2 MB)
    float* ws   = (float*)d_ws;
    float* pmax = ws;
    float* psum = pmax + (size_t)2 * BT * VT;
    float* tgtl = psum + (size_t)2 * BT * VT;
    float* logp = tgtl + (size_t)2 * BT;

    dim3 g1(MT2, VT, 2);  // M fastest so blocks sharing a W tile co-run -> W streams ~once
    kto_gemm_lse<<<g1, dim3(256), 0, stream>>>(x, w, bs, rx, rw, rb, tg, pmax, psum, tgtl);

    kto_combine<<<dim3((2 * BT + 255) / 256), dim3(256), 0, stream>>>(pmax, psum, tgtl, logp);

    kto_finalize<<<dim3(1), dim3(64), 0, stream>>>(logp, tg, pl, kl, (float*)d_out);
}
